// Generator_9019431321808
// MI455X (gfx1250) — matrix operand order repaired and checked by reading, not compiled, not run on hardware
//
#include <hip/hip_runtime.h>
#include <hip/hip_bf16.h>

typedef __attribute__((ext_vector_type(8)))  _Float16 v8h;
typedef __attribute__((ext_vector_type(16))) _Float16 v16h;
typedef __attribute__((ext_vector_type(8)))  float    v8f;

#define B_  4096
#define E_  128
#define H_  512
#define V_  50257
#define L_  16
#define KX  160      /* E_+L_ padded up to a multiple of 32 */
#define G3  1536     /* 3*H_ */

#define SMAX_T 1024                      /* threads per softmax block (32 waves) */
#define NPT    50                        /* ceil(V_ / SMAX_T) values per thread  */

// ---------------- conversion / gather kernels ----------------

__global__ void build_x_kernel(const int* __restrict__ inp,
                               const float* __restrict__ labels,
                               const float* __restrict__ emb,
                               _Float16* __restrict__ x) {
  int i = blockIdx.x * blockDim.x + threadIdx.x;
  if (i >= B_ * KX) return;
  int b = i / KX, c = i % KX;
  float v;
  if (c < E_)            v = emb[(size_t)inp[b] * E_ + c];
  else if (c < E_ + L_)  v = labels[b * L_ + (c - E_)];
  else                   v = 0.0f;
  x[i] = (_Float16)v;
}

__global__ void pad_wih_kernel(const float* __restrict__ w,
                               _Float16* __restrict__ o) {
  int i = blockIdx.x * blockDim.x + threadIdx.x;
  if (i >= G3 * KX) return;
  int n = i / KX, c = i % KX;
  o[i] = (_Float16)((c < E_ + L_) ? w[n * (E_ + L_) + c] : 0.0f);
}

__global__ void f32_to_f16_kernel(const float* __restrict__ in,
                                  _Float16* __restrict__ out, size_t n) {
  size_t i = (size_t)blockIdx.x * blockDim.x + threadIdx.x;
  if (i < n) out[i] = (_Float16)in[i];
}

// ---------------- WMMA GEMM: C[M,N] = A[M,K] * W[N,K]^T + bias[N] ----------------
// A fragment (16x32 f16, ISA 7.12.2): lane l holds row (l&15); K chunks at
// k0 + (l>>4)*8 and k0 + (l>>4)*8 + 16 (8 halves each).
__device__ __forceinline__ v16h load_a_frag(const _Float16* __restrict__ A,
                                            int ld, int m0, int k0, int lane) {
  int grp = lane >> 4, r = lane & 15;
  const _Float16* p = A + (size_t)(m0 + r) * ld + k0 + grp * 8;
  v8h c0 = *(const v8h*)p;
  v8h c1 = *(const v8h*)(p + 16);
  return __builtin_shufflevector(c0, c1, 0,1,2,3,4,5,6,7,8,9,10,11,12,13,14,15);
}

// B fragment (32x16 f16): lane l holds column (l&15) of B == row (n0 + (l&15))
// of W; K chunks at k0 + (l>>4)*8 and k0 + (l>>4)*8 + 16 (8 halves each).
__device__ __forceinline__ v16h load_b_frag(const _Float16* __restrict__ W,
                                            int ld, int n0, int k0, int lane, int N) {
  int grp = lane >> 4, r = lane & 15;
  int row = n0 + r; if (row >= N) row = N - 1;   // clamp keeps EXEC all-ones
  const _Float16* p = W + (size_t)row * ld + k0 + grp * 8;
  v8h c0 = *(const v8h*)p;
  v8h c1 = *(const v8h*)(p + 16);
  return __builtin_shufflevector(c0, c1, 0,1,2,3,4,5,6,7,8,9,10,11,12,13,14,15);
}

__global__ __launch_bounds__(256)
void wmma_gemm_bias(const _Float16* __restrict__ A,   // M x K (row major)
                    const _Float16* __restrict__ W,   // N x K (row major)
                    const float*    __restrict__ bias,// N
                    float*          __restrict__ C,   // M x N (row major)
                    int M, int N, int K) {
  int lane = threadIdx.x & 31;
  int wave = threadIdx.x >> 5;          // 8 waves: 2 (M) x 4 (N)
  int wm = wave & 1, wn = wave >> 1;
  int m0 = blockIdx.y * 64  + wm * 32;
  int n0 = blockIdx.x * 128 + wn * 32;

  v8f acc00 = {}, acc01 = {}, acc10 = {}, acc11 = {};
  for (int k0 = 0; k0 < K; k0 += 32) {
    v16h a0 = load_a_frag(A, K, m0,      k0, lane);
    v16h a1 = load_a_frag(A, K, m0 + 16, k0, lane);
    v16h b0 = load_b_frag(W, K, n0,      k0, lane, N);
    v16h b1 = load_b_frag(W, K, n0 + 16, k0, lane, N);
    acc00 = __builtin_amdgcn_wmma_f32_16x16x32_f16(false, a0, false, b0, (short)0, acc00, false, false);
    acc01 = __builtin_amdgcn_wmma_f32_16x16x32_f16(false, a0, false, b1, (short)0, acc01, false, false);
    acc10 = __builtin_amdgcn_wmma_f32_16x16x32_f16(false, a1, false, b0, (short)0, acc10, false, false);
    acc11 = __builtin_amdgcn_wmma_f32_16x16x32_f16(false, a1, false, b1, (short)0, acc11, false, false);
  }

  // C/D layout: lanes 0-15 -> N=lane, M rows m0..m0+7; lanes 16-31 -> M rows +8.
  int grp = lane >> 4, col = lane & 15;
  v8f accs[2][2] = {{acc00, acc01}, {acc10, acc11}};
  for (int ti = 0; ti < 2; ++ti) {
    for (int tj = 0; tj < 2; ++tj) {
      int n = n0 + tj * 16 + col;
      if (n < N) {
        float bv = bias[n];
        int mb = m0 + ti * 16 + grp * 8;
#pragma unroll
        for (int r = 0; r < 8; ++r)
          C[(size_t)(mb + r) * N + n] = accs[ti][tj][r] + bv;
      }
    }
  }
}

// ---------------- GRU gates (biases already folded into gi/gh) ----------------

__global__ void gru_gate_kernel(const float* __restrict__ gi,
                                const float* __restrict__ gh,
                                const float* __restrict__ h0,
                                float* __restrict__ hnew_f32,
                                _Float16* __restrict__ hnew_f16) {
  int i = blockIdx.x * blockDim.x + threadIdx.x;
  if (i >= B_ * H_) return;
  int b = i / H_, j = i % H_;
  size_t base = (size_t)b * G3 + j;
  float ir = gi[base], iz = gi[base + H_], in_ = gi[base + 2 * H_];
  float hr = gh[base], hz = gh[base + H_], hn  = gh[base + 2 * H_];
  float r = 1.0f / (1.0f + __expf(-(ir + hr)));
  float z = 1.0f / (1.0f + __expf(-(iz + hz)));
  float n = tanhf(in_ + r * hn);
  float h = h0[i];
  float o = (1.0f - z) * n + z * h;
  hnew_f32[i] = o;
  hnew_f16[i] = (_Float16)o;
}

// ---------------- fused one-read log_softmax (row register-resident) ----------------
// One block per row; 1024 threads x 50 regs hold all 50257 values. HBM traffic
// is 1 read + 1 write instead of 2 reads + 1 write.

__global__ __launch_bounds__(SMAX_T)
void log_softmax_fused_kernel(float* __restrict__ out, int N) {
  __shared__ float sm[32], ss[32];
  float* row = out + (size_t)blockIdx.x * N;
  int tid = threadIdx.x;
  const float ninf = -__builtin_inff();

  float v[NPT];
#pragma unroll
  for (int i = 0; i < NPT; ++i) {
    int c = tid + i * SMAX_T;
    v[i] = (c < N) ? row[c] : ninf;
  }

  // per-thread max, then sum of exp (each thread has >=1 finite value: c=tid<N)
  float m = ninf;
#pragma unroll
  for (int i = 0; i < NPT; ++i) m = fmaxf(m, v[i]);
  float s = 0.0f;
#pragma unroll
  for (int i = 0; i < NPT; ++i) s += __expf(v[i] - m);   // exp(-inf)=0 for pads

  // wave32 reduction
#pragma unroll
  for (int off = 16; off > 0; off >>= 1) {
    float m2 = __shfl_xor(m, off, 32);
    float s2 = __shfl_xor(s, off, 32);
    float nm = fmaxf(m, m2);
    s = s * __expf(m - nm) + s2 * __expf(m2 - nm);
    m = nm;
  }
  int wave = tid >> 5, lane = tid & 31;
  if (lane == 0) { sm[wave] = m; ss[wave] = s; }
  __syncthreads();
  if (wave == 0) {
    float mm = sm[lane];   // 32 waves -> all 32 lanes valid
    float sw = ss[lane];
#pragma unroll
    for (int off = 16; off > 0; off >>= 1) {
      float m2 = __shfl_xor(mm, off, 32);
      float s2 = __shfl_xor(sw, off, 32);
      float nm = fmaxf(mm, m2);
      sw = sw * __expf(mm - nm) + s2 * __expf(m2 - nm);
      mm = nm;
    }
    if (lane == 0) { sm[0] = mm; ss[0] = sw; }
  }
  __syncthreads();
  float lse = sm[0] + __logf(ss[0]);

#pragma unroll
  for (int i = 0; i < NPT; ++i) {
    int c = tid + i * SMAX_T;
    if (c < N) __builtin_nontemporal_store(v[i] - lse, &row[c]);
  }
}

// ---------------- launch ----------------

extern "C" void kernel_launch(void* const* d_in, const int* in_sizes, int n_in,
                              void* d_out, int out_size, void* d_ws, size_t ws_size,
                              hipStream_t stream) {
  (void)in_sizes; (void)n_in; (void)out_size; (void)ws_size;
  const int*   inp    = (const int*)  d_in[0];
  const float* hidden = (const float*)d_in[1];
  const float* labels = (const float*)d_in[2];
  const float* emb    = (const float*)d_in[3];
  const float* W_ih   = (const float*)d_in[4];
  const float* W_hh   = (const float*)d_in[5];
  const float* b_ih   = (const float*)d_in[6];
  const float* b_hh   = (const float*)d_in[7];
  const float* W_out  = (const float*)d_in[8];
  const float* b_out  = (const float*)d_in[9];

  char* ws = (char*)d_ws;
  size_t off = 0;
  auto alloc = [&](size_t bytes) -> void* {
    void* p = ws + off;
    off = (off + bytes + 255) & ~(size_t)255;
    return p;
  };
  _Float16* x_f16    = (_Float16*)alloc((size_t)B_ * KX * 2);
  _Float16* wih_f16  = (_Float16*)alloc((size_t)G3 * KX * 2);
  _Float16* h_f16    = (_Float16*)alloc((size_t)B_ * H_ * 2);
  _Float16* whh_f16  = (_Float16*)alloc((size_t)G3 * H_ * 2);
  _Float16* hnew_f16 = (_Float16*)alloc((size_t)B_ * H_ * 2);
  _Float16* wout_f16 = (_Float16*)alloc((size_t)V_ * H_ * 2);
  float*    gi       = (float*)   alloc((size_t)B_ * G3 * 4);
  float*    gh       = (float*)   alloc((size_t)B_ * G3 * 4);

  float* out      = (float*)d_out;                 // [B, V]
  float* hnew_out = out + (size_t)B_ * V_;         // [1, B, H]

  const int t = 256;
  build_x_kernel<<<(B_ * KX + t - 1) / t, t, 0, stream>>>(inp, labels, emb, x_f16);
  pad_wih_kernel<<<(G3 * KX + t - 1) / t, t, 0, stream>>>(W_ih, wih_f16);
  f32_to_f16_kernel<<<(int)(((size_t)B_ * H_ + t - 1) / t), t, 0, stream>>>(hidden, h_f16, (size_t)B_ * H_);
  f32_to_f16_kernel<<<(int)(((size_t)G3 * H_ + t - 1) / t), t, 0, stream>>>(W_hh, whh_f16, (size_t)G3 * H_);
  f32_to_f16_kernel<<<(int)(((size_t)V_ * H_ + t - 1) / t), t, 0, stream>>>(W_out, wout_f16, (size_t)V_ * H_);

  // gi = x @ W_ih^T + b_ih   (K padded to 160)
  wmma_gemm_bias<<<dim3((G3 + 127) / 128, B_ / 64), t, 0, stream>>>(
      x_f16, wih_f16, b_ih, gi, B_, G3, KX);
  // gh = h @ W_hh^T + b_hh
  wmma_gemm_bias<<<dim3((G3 + 127) / 128, B_ / 64), t, 0, stream>>>(
      h_f16, whh_f16, b_hh, gh, B_, G3, H_);

  gru_gate_kernel<<<(B_ * H_ + t - 1) / t, t, 0, stream>>>(gi, gh, hidden, hnew_out, hnew_f16);

  // logits = h_new @ W_out^T + b_out  -> d_out (dominant GEMM, W_out f16 L2-resident)
  wmma_gemm_bias<<<dim3((V_ + 127) / 128, B_ / 64), t, 0, stream>>>(
      hnew_f16, wout_f16, b_out, out, B_, V_, H_);

  log_softmax_fused_kernel<<<B_, SMAX_T, 0, stream>>>(out, V_);
}
